// LocalEncoderSDESepPara2_39101382263115
// MI455X (gfx1250) — matrix operand order repaired and checked by reading, not compiled, not run on hardware
//
#include <hip/hip_runtime.h>
#include <hip/hip_bf16.h>
#include <cstdint>

// Problem constants
#define NB    16
#define NC    128
#define NTP   11
#define NPN   1024
#define NSTEPS 21

typedef __attribute__((ext_vector_type(16))) _Float16 v16h;
typedef __attribute__((ext_vector_type(8)))  _Float16 v8h;
typedef __attribute__((ext_vector_type(8)))  float    v8f;
typedef __attribute__((ext_vector_type(4)))  unsigned int v4u;
typedef __attribute__((ext_vector_type(8)))  int v8i;
typedef __attribute__((ext_vector_type(4)))  int v4i;

// ws blob layout (bytes):
//   [0,2560)        : biases f32, order bf,bg,bz,br,bh (5*128)
//   [2560,264704)   : 8 weight matrices as f16 WMMA A-fragments, 32768B each
//                     order: Wf,Wg,Wz,Uz,Wr,Ur,Wh,Uh
#define WS_BIAS_BYTES  2560
#define WS_BLOB_BYTES  264704
#define TDM_ROW_DW     16544u

// LDS layout (bytes):
#define LDS_W_OFF      2560                  // f16 fragments start
#define LDS_STAGE_OFF  264704                // per-wave staging
#define LDS_STAGE_PER_WAVE 12288             // 4KB h[16][128] + 8KB xh[16][256]
#define LDS_TOTAL      (LDS_STAGE_OFF + 4*LDS_STAGE_PER_WAVE)  // 313856

// Output offsets (floats) in d_out: (drift_loc, all_drift_loc, all_diff_z)
#define OFF_ALL  23068672ULL                 // 16*11*128*1024
#define OFF_DZ   67108864ULL                 // OFF_ALL + 16*21*128*1024

__device__ __forceinline__ v8f wmma_f16(v16h a, v16h b, v8f c) {
    return __builtin_amdgcn_wmma_f32_16x16x32_f16(false, a, false, b,
                                                  (short)0, c, false, false);
}

// Branchless fast transcendentals: v_exp_f32 (2^x) + v_rcp_f32.
#define LOG2E 1.4426950408889634f
__device__ __forceinline__ float sigm(float x) {
    return __builtin_amdgcn_rcpf(1.0f + __builtin_amdgcn_exp2f(-LOG2E * x));
}
__device__ __forceinline__ float tanh_fast(float x) {
    // tanh(x) = 1 - 2*sigmoid(-2x)
    return 1.0f - 2.0f * __builtin_amdgcn_rcpf(1.0f + __builtin_amdgcn_exp2f(2.0f * LOG2E * x));
}

// ------------------------------------------------------------------
// Prep: swizzle 8 weight matrices into WMMA A-fragment order (f16),
// copy biases, all into the ws blob.
// A-fragment order used here (contiguous k, same as the B loads):
//   M = mt*16 + (lane&15)
//   K = kt*32 + (lane<16?0:16) + h
// ------------------------------------------------------------------
__global__ void sde_prep_kernel(const float* __restrict__ Wf, const float* __restrict__ Wg,
                                const float* __restrict__ Wz, const float* __restrict__ Uz,
                                const float* __restrict__ Wr, const float* __restrict__ Ur,
                                const float* __restrict__ Wh, const float* __restrict__ Uh,
                                const float* __restrict__ bf, const float* __restrict__ bg,
                                const float* __restrict__ bz, const float* __restrict__ br,
                                const float* __restrict__ bh, char* __restrict__ ws)
{
    int tid = blockIdx.x * 256 + threadIdx.x;
    if (tid < 640) {
        const float* bp;
        int which = tid >> 7, i = tid & 127;
        switch (which) {
            case 0:  bp = bf; break;
            case 1:  bp = bg; break;
            case 2:  bp = bz; break;
            case 3:  bp = br; break;
            default: bp = bh; break;
        }
        ((float*)ws)[tid] = bp[i];
        return;
    }
    int t = tid - 640;
    if (t >= 8 * 16384) return;
    int m = t >> 14, j = t & 16383;
    const float* W;
    switch (m) {
        case 0: W = Wf; break; case 1: W = Wg; break;
        case 2: W = Wz; break; case 3: W = Uz; break;
        case 4: W = Wr; break; case 5: W = Ur; break;
        case 6: W = Wh; break; default: W = Uh; break;
    }
    int mt = j >> 11, kt = (j >> 9) & 3, lane = (j >> 4) & 31, h = j & 15;
    int v = h >> 1, e = h & 1;
    int K = kt * 32 + ((lane < 16) ? 0 : 16) + h;
    int M = mt * 16 + (lane & 15);
    _Float16* wsH = (_Float16*)(ws + WS_BIAS_BYTES);
    wsH[m * 16384 + j] = (_Float16)W[M * 128 + K];
}

// ------------------------------------------------------------------
// Main scan kernel: grid = 256 (16 batches x 16 column-groups of 64),
// block = 128 threads = 4 waves; each wave owns 16 columns for all steps.
// ------------------------------------------------------------------
__global__ __launch_bounds__(128, 1)
void sde_scan_kernel(const float* __restrict__ aa, const float* __restrict__ noise,
                     const float* __restrict__ hidden, const char* __restrict__ ws,
                     float* __restrict__ out)
{
    extern __shared__ __align__(32) char smem[];
    const int lane   = threadIdx.x & 31;
    const int wv     = threadIdx.x >> 5;          // wave in block, 0..3
    const int b      = blockIdx.x >> 4;           // batch
    const int grp    = blockIdx.x & 15;           // column group
    const int cl     = lane & 15;                 // local column within wave
    const int hiHalf = (lane >= 16) ? 1 : 0;
    const int col    = grp * 64 + wv * 16 + cl;   // global p index

    float*    biasL = (float*)smem;                                   // 640 f32
    _Float16* Wl    = (_Float16*)(smem + LDS_W_OFF);                  // fragments
    _Float16* stH   = (_Float16*)(smem + LDS_STAGE_OFF + wv * LDS_STAGE_PER_WAVE);        // [16][128]
    _Float16* stXH  = (_Float16*)(smem + LDS_STAGE_OFF + wv * LDS_STAGE_PER_WAVE + 4096); // [16][256]

    // ---- TDM: pull the whole weight/bias blob (259KB) into LDS at offset 0 ----
    if (wv == 0) {
        unsigned long long ga = (unsigned long long)(uintptr_t)ws;
        v4u g0;
        g0.x = 1u;                                             // count=1 (valid D#)
        g0.y = 0u;                                             // lds_addr = 0
        g0.z = (unsigned)(ga & 0xFFFFFFFFu);                   // global_addr[31:0]
        g0.w = (unsigned)((ga >> 32) & 0x1FFFFFFu) | (2u << 30); // addr[56:32] | type=2
        v8i g1;
        g1[0] = (int)(2u << 16);          // data_size = 4B
        g1[1] = (int)(TDM_ROW_DW << 16);  // tensor_dim0[15:0]
        g1[2] = (int)(4u << 16);          // tensor_dim0 hi=0 | tensor_dim1[15:0]=4
        g1[3] = (int)(TDM_ROW_DW << 16);  // tensor_dim1 hi=0 | tile_dim0
        g1[4] = 4;                        // tile_dim1=4, tile_dim2=0
        g1[5] = (int)TDM_ROW_DW;          // tensor_dim0_stride lo32
        g1[6] = 0; g1[7] = 0;
        v4i g2 = {0, 0, 0, 0}, g3 = {0, 0, 0, 0};
#if __has_include(<hip/amd_detail/amd_gfx1250_TDM.h>)
        v8i g4 = {0, 0, 0, 0, 0, 0, 0, 0};
        __builtin_amdgcn_tensor_load_to_lds(g0, g1, g2, g3, g4, 0);
#else
        __builtin_amdgcn_tensor_load_to_lds(g0, g1, g2, g3, 0);
#endif
        __builtin_amdgcn_s_wait_tensorcnt(0);
    }
    __syncthreads();

    // ---- h0 staging: stH[cl][c] = hidden[c], column-major f16 ----
    for (int i = 0; i < 8; ++i) {
        int c0 = hiHalf * 64 + i * 8;
        v8h pk;
        for (int r = 0; r < 8; ++r) pk[r] = (_Float16)hidden[c0 + r];
        *(v8h*)&stH[cl * 128 + c0] = pk;
    }

    for (int s = 0; s < NSTEPS; ++s) {
        const float dt  = (s == 0) ? 0.01f : -0.5f;
        const float sqa = (s == 0) ? 0.1f  : 0.70710678f;
        const size_t inBase = (((size_t)s * NB + b) * NC) * NPN + col;

        if (s + 1 < NSTEPS)
            __builtin_prefetch(&aa[inBase + (size_t)NB * NC * NPN], 0, 3);

        // ---- stage x_t into stXH rows [0,128) (column-major f16), NT loads ----
        for (int i = 0; i < 32; ++i) {
            int c0 = i * 4 + hiHalf * 2;
            float x0 = __builtin_nontemporal_load(&aa[inBase + (size_t)c0 * NPN]);
            float x1 = __builtin_nontemporal_load(&aa[inBase + (size_t)(c0 + 1) * NPN]);
            union { _Float16 h2[2]; unsigned u; } uu;
            uu.h2[0] = (_Float16)x0; uu.h2[1] = (_Float16)x1;
            *(unsigned*)&stXH[cl * 256 + c0] = uu.u;
        }

        // ---- phase 1: drift/diffusion from h, form h_sde, emit dz ----
        for (int mt = 0; mt < 8; ++mt) {
            v8f a1 = {0, 0, 0, 0, 0, 0, 0, 0};
            v8f a2 = {0, 0, 0, 0, 0, 0, 0, 0};
            for (int kt = 0; kt < 4; ++kt) {
                v16h bh = *(const v16h*)&stH[cl * 128 + kt * 32 + hiHalf * 16];
                int fo = ((mt * 4 + kt) * 32 + lane) * 16;
                v16h af = *(const v16h*)&Wl[0 * 16384 + fo];
                v16h ag = *(const v16h*)&Wl[1 * 16384 + fo];
                a1 = wmma_f16(af, bh, a1);
                a2 = wmma_f16(ag, bh, a2);
            }
            int rowBase = mt * 16 + hiHalf * 8;
            v8h hv = *(const v8h*)&stH[cl * 128 + rowBase];
            const size_t cBase  = inBase + (size_t)rowBase * NPN;
            const size_t ozBase = OFF_DZ + ((((size_t)b * NSTEPS + s) * NC) + rowBase) * NPN + col;
            v8h hsdeH;
            for (int r = 0; r < 8; ++r) {
                float drift = tanh_fast(a1[r] + biasL[0 * 128 + rowBase + r]);
                float diff  = sigm(a2[r] + biasL[1 * 128 + rowBase + r]);
                float nz = __builtin_nontemporal_load(&noise[cBase + (size_t)r * NPN]);
                float dz = diff * sqa * nz;
                float hs = (float)hv[r] + drift * dt + dz;
                __builtin_nontemporal_store(dz, &out[ozBase + (size_t)r * NPN]);
                hsdeH[r] = (_Float16)hs;
            }
            *(v8h*)&stXH[cl * 256 + 128 + rowBase] = hsdeH;  // rows [128,256) = h_sde
        }

        // ---- phase 2: z, r, n from [x ; h_sde], produce h_new ----
        for (int mt = 0; mt < 8; ++mt) {
            v8f az = {0, 0, 0, 0, 0, 0, 0, 0};
            v8f ar = {0, 0, 0, 0, 0, 0, 0, 0};
            v8f an = {0, 0, 0, 0, 0, 0, 0, 0};
            v8f am = {0, 0, 0, 0, 0, 0, 0, 0};
            for (int kt = 0; kt < 4; ++kt) {
                v16h bx  = *(const v16h*)&stXH[cl * 256 + kt * 32 + hiHalf * 16];
                v16h bs2 = *(const v16h*)&stXH[cl * 256 + 128 + kt * 32 + hiHalf * 16];
                int fo = ((mt * 4 + kt) * 32 + lane) * 16;
                v16h wz = *(const v16h*)&Wl[2 * 16384 + fo];
                v16h uz = *(const v16h*)&Wl[3 * 16384 + fo];
                v16h wr = *(const v16h*)&Wl[4 * 16384 + fo];
                v16h ur = *(const v16h*)&Wl[5 * 16384 + fo];
                v16h wh = *(const v16h*)&Wl[6 * 16384 + fo];
                v16h uh = *(const v16h*)&Wl[7 * 16384 + fo];
                az = wmma_f16(wz, bx,  az);
                az = wmma_f16(uz, bs2, az);
                ar = wmma_f16(wr, bx,  ar);
                ar = wmma_f16(ur, bs2, ar);
                an = wmma_f16(wh, bx,  an);
                am = wmma_f16(uh, bs2, am);
            }
            int rowBase = mt * 16 + hiHalf * 8;
            v8h hsdeH = *(const v8h*)&stXH[cl * 256 + 128 + rowBase];
            const size_t oA = OFF_ALL + ((((size_t)b * NSTEPS + s) * NC) + rowBase) * NPN + col;
            v8h hnewH;
            float hnv[8];
            for (int r = 0; r < 8; ++r) {
                float z  = sigm(az[r] + biasL[2 * 128 + rowBase + r]);
                float rr = sigm(ar[r] + biasL[3 * 128 + rowBase + r]);
                float n  = tanh_fast(an[r] + biasL[4 * 128 + rowBase + r] + rr * am[r]);
                float hs = (float)hsdeH[r];
                float hn = (1.0f - z) * n + z * hs;
                __builtin_nontemporal_store(hn, &out[oA + (size_t)r * NPN]);
                hnv[r] = hn;
                hnewH[r] = (_Float16)hn;
            }
            if ((s & 1) == 0) {
                const size_t oD = ((((size_t)b * NTP + (s >> 1)) * NC) + rowBase) * NPN + col;
                for (int r = 0; r < 8; ++r)
                    __builtin_nontemporal_store(hnv[r], &out[oD + (size_t)r * NPN]);
            }
            *(v8h*)&stH[cl * 128 + rowBase] = hnewH;   // h for next step
        }
    }
}

extern "C" void kernel_launch(void* const* d_in, const int* in_sizes, int n_in,
                              void* d_out, int out_size, void* d_ws, size_t ws_size,
                              hipStream_t stream) {
    (void)in_sizes; (void)n_in; (void)out_size; (void)ws_size;
    const float* aa     = (const float*)d_in[1];
    const float* noise  = (const float*)d_in[2];
    const float* hidden = (const float*)d_in[3];

    sde_prep_kernel<<<515, 256, 0, stream>>>(
        (const float*)d_in[4],  (const float*)d_in[6],
        (const float*)d_in[8],  (const float*)d_in[9],
        (const float*)d_in[11], (const float*)d_in[12],
        (const float*)d_in[14], (const float*)d_in[15],
        (const float*)d_in[5],  (const float*)d_in[7],
        (const float*)d_in[10], (const float*)d_in[13],
        (const float*)d_in[16], (char*)d_ws);

    sde_scan_kernel<<<NB * (NPN / 64), 128, LDS_TOTAL, stream>>>(
        aa, noise, hidden, (const char*)d_ws, (float*)d_out);
}
